// LSTM_47167330845117
// MI455X (gfx1250) — compile-verified, matrix operand order checked by reading, not run on hardware
//
#include <hip/hip_runtime.h>
#include <stdint.h>

#define N_BATCH 256
#define T_STEPS 512
#define D_IN    256
#define H_DIM   256
#define G4H     1024   // 4*H
#define K_TOT   512    // D + H
#define KT_N    16     // K_TOT / 32
#define NT_N    64     // G4H / 16

typedef __bf16 v16bf __attribute__((ext_vector_type(16)));
typedef float  v8f   __attribute__((ext_vector_type(8)));
typedef float  v4f   __attribute__((ext_vector_type(4)));

// fp32 -> bf16 round-to-nearest-even (bit pattern)
__device__ __forceinline__ uint16_t f2bf(float f) {
    uint32_t u = __float_as_uint(f);
    uint32_t r = u + 0x7FFFu + ((u >> 16) & 1u);
    return (uint16_t)(r >> 16);
}

// CDNA5 16-bit 16x32 fragment layout (ISA 7.12.2):
//   lane = row + 16 * K[3];  slot(half index 0..15) = (K & 7) + 8 * (K >= 16)
__device__ __forceinline__ int frag_slot(int K) { return (K & 7) + ((K & 16) ? 8 : 0); }

// branchless fast tanh: sign(x) * (1 - e) / (1 + e), e = exp(-2|x|)
__device__ __forceinline__ float fast_tanh(float v) {
    float a = __builtin_fabsf(v);
    float e = __expf(-2.0f * a);
    float r = (1.0f - e) / (1.0f + e);
    return __builtin_copysignf(r, v);
}
__device__ __forceinline__ float fast_sigmoid(float v) {
    return 1.0f / (1.0f + __expf(-v));
}

// ---------------------------------------------------------------------------
// One-shot: pack [Wx; Wh] (512 x 1024) as bf16 in B-fragment order:
//   wpack[((kt*64 + ntile)*32 + lane)*16 + slot]
// ---------------------------------------------------------------------------
__global__ __launch_bounds__(256) void pack_weights(
    const float* __restrict__ wii, const float* __restrict__ whi,
    const float* __restrict__ wif, const float* __restrict__ whf,
    const float* __restrict__ wig, const float* __restrict__ whg,
    const float* __restrict__ wio, const float* __restrict__ who,
    uint16_t* __restrict__ wpack)
{
    int idx = blockIdx.x * blockDim.x + threadIdx.x;
    if (idx >= K_TOT * G4H) return;
    int slot  = idx & 15;
    int lane  = (idx >> 4) & 31;
    int ntile = (idx >> 9) & 63;
    int kt    = idx >> 15;
    int half  = lane >> 4;
    int nl    = lane & 15;
    int K   = (slot & 7) + 8 * half + ((slot >= 8) ? 16 : 0);
    int k   = kt * 32 + K;          // 0..511 : [Wx rows | Wh rows]
    int col = ntile * 16 + nl;      // 0..1023: gate-major columns (i,f,g,o)
    int gate = col >> 8;
    int hc   = col & 255;
    const float* wx = (gate == 0) ? wii : (gate == 1) ? wif : (gate == 2) ? wig : wio;
    const float* wh = (gate == 0) ? whi : (gate == 1) ? whf : (gate == 2) ? whg : who;
    float v = (k < D_IN) ? wx[k * H_DIM + hc] : wh[(k - D_IN) * H_DIM + hc];
    wpack[idx] = f2bf(v);
}

__global__ __launch_bounds__(256) void pack_bias(
    const float* __restrict__ bii, const float* __restrict__ bhi,
    const float* __restrict__ bif, const float* __restrict__ bhf,
    const float* __restrict__ big, const float* __restrict__ bhg,
    const float* __restrict__ bio, const float* __restrict__ bho,
    float* __restrict__ bias)
{
    int j = blockIdx.x * blockDim.x + threadIdx.x;
    if (j >= G4H) return;
    int gate = j >> 8, hc = j & 255;
    const float* bx = (gate == 0) ? bii : (gate == 1) ? bif : (gate == 2) ? big : bio;
    const float* bh = (gate == 0) ? bhi : (gate == 1) ? bhf : (gate == 2) ? bhg : bho;
    bias[j] = bx[hc] + bh[hc];
}

// ---------------------------------------------------------------------------
// Persistent LSTM: 16 blocks x 512 threads (16 waves = 4 per SIMD). Block b
// owns batch rows [16b, 16b+16). Per step:
//   stage x_t -> LDS A (bf16 fragment layout)
//   fused GEMM [x|h](16x512) @ W(512x1024), 4 N-tiles/wave; kt fully unrolled
//   with an explicit 2-stage B pipeline (b0/b1 alternate by kt parity — pure
//   SSA renames after unrolling) so each WMMA group runs with ~8 loads in
//   flight instead of serializing on L2 latency
//   gate exchange via transposed LDS layout g[col][row] (b128 stores/loads)
//   elementwise cell update (c,h in registers), h -> A h-half in place
// ---------------------------------------------------------------------------
__global__ __launch_bounds__(512)
__attribute__((amdgpu_waves_per_eu(4)))
void lstm_persistent(
    const float* __restrict__ x, const uint16_t* __restrict__ wpack,
    const float* __restrict__ bias, float* __restrict__ out)
{
    extern __shared__ char smem[];
    uint16_t* a_sh = (uint16_t*)smem;                       // [KT_N][32][16] bf16 = 16 KB
    float*    g_sh = (float*)(smem + KT_N * 32 * 16 * 2);   // [1024 cols][16 rows] = 64 KB

    const int tid  = threadIdx.x;           // 0..511
    const int wave = tid >> 5;              // 0..15
    const int lane = tid & 31;
    const int n0   = blockIdx.x * 16;

    // elementwise ownership: column j, rows [rb, rb+8)
    const int j  = tid & 255;
    const int rb = (tid >> 8) * 8;          // 0 or 8

    float c_reg[8], h_reg[8];
#pragma unroll
    for (int r = 0; r < 8; ++r) { c_reg[r] = 0.f; h_reg[r] = 0.f; }

    // zero A (h-half must start at 0; x-half overwritten every step)
    for (int i = tid; i < KT_N * 32 * 16; i += 512) a_sh[i] = 0;

    const float b0s = bias[j];
    const float b1s = bias[256 + j];
    const float b2s = bias[512 + j];
    const float b3s = bias[768 + j];

    // A-staging addressing: this thread feeds column k=j (x half), k=256+j (h half)
    const int Kc      = j & 31;
    const int kt_x    = j >> 5;             // 0..7
    const int kt_h    = 8 + (j >> 5);       // 8..15
    const int lane_hi = ((Kc >> 3) & 1) ? 16 : 0;
    const int slot_c  = frag_slot(Kc);

    // per-wave weight base: fragment (kt, nt) at constant byte offset from here
    const int ntg0 = wave * 4;
    const uint16_t* wbase = wpack + ((size_t)ntg0 * 32 + lane) * 16;

    __syncthreads();

    for (int t = 0; t < T_STEPS; ++t) {
        // ---- stage x_t tile (rows rb..rb+7 of 16, col j) f32 -> bf16 frag layout
#pragma unroll
        for (int mm = 0; mm < 8; ++mm) {
            const int m = rb + mm;
            const float* p = x + (size_t)(n0 + m) * (T_STEPS * D_IN)
                               + (size_t)t * D_IN + j;
            float v = *p;
            if (t + 1 < T_STEPS) __builtin_prefetch(p + D_IN, 0, 1);
            a_sh[(kt_x * 32 + (m + lane_hi)) * 16 + slot_c] = f2bf(v);
        }
        __syncthreads();

        // opaque zero offset: re-materialized every step so LICM cannot hoist
        // the (invariant) weight loads out of the t loop and spill to scratch;
        // keeps wbase's global address space -> global_load_b128
        int woff = 0;
        asm volatile("" : "+v"(woff));
        const uint16_t* wb = wbase + (uint32_t)woff;

        // ---- GEMM: gates(16x1024) = A(16x512) @ W(512x1024); wave owns 4 N-tiles
        // 2-stage B pipeline over fully-unrolled kt: loads of kt+1 precede
        // the WMMA group of kt in program order.
        v8f acc[4] = {};
        v16bf bp0[4], bp1[4];
#pragma unroll
        for (int nt = 0; nt < 4; ++nt)
            bp0[nt] = *(const v16bf*)(wb + (size_t)nt * (32 * 16));

#pragma unroll
        for (int kt = 0; kt < KT_N; ++kt) {
            v16bf (&cur)[4] = (kt & 1) ? bp1 : bp0;
            v16bf (&nxt)[4] = (kt & 1) ? bp0 : bp1;
            if (kt + 1 < KT_N) {
#pragma unroll
                for (int nt = 0; nt < 4; ++nt)
                    nxt[nt] = *(const v16bf*)(wb + (size_t)(kt + 1) * (NT_N * 32 * 16)
                                                 + (size_t)nt * (32 * 16));
            }
            v16bf af = *(const v16bf*)(a_sh + (kt * 32 + lane) * 16);
#pragma unroll
            for (int nt = 0; nt < 4; ++nt)
                acc[nt] = __builtin_amdgcn_wmma_f32_16x16x32_bf16(
                    false, af, false, cur[nt], (short)0, acc[nt], false, false);
        }

        // ---- scatter accumulators to transposed gate LDS: g_sh[col*16 + row]
        // C/D layout: lane<16 holds rows 0-7 (v=row), lane>=16 rows 8-15
        const int m_hi = (lane >= 16) ? 8 : 0;
        const int nl   = lane & 15;
#pragma unroll
        for (int nt = 0; nt < 4; ++nt) {
            const int col = (ntg0 + nt) * 16 + nl;
            float* gp = g_sh + col * 16 + m_hi;
            v4f lo = { acc[nt][0], acc[nt][1], acc[nt][2], acc[nt][3] };
            v4f hi = { acc[nt][4], acc[nt][5], acc[nt][6], acc[nt][7] };
            *(v4f*)(gp)     = lo;   // ds_store_b128
            *(v4f*)(gp + 4) = hi;   // ds_store_b128
        }
        __syncthreads();

        // ---- elementwise cell update; thread owns column j, rows rb..rb+7
        v8f gi = *(const v8f*)(g_sh + (0 * 256 + j) * 16 + rb);
        v8f gf = *(const v8f*)(g_sh + (1 * 256 + j) * 16 + rb);
        v8f gg = *(const v8f*)(g_sh + (2 * 256 + j) * 16 + rb);
        v8f go = *(const v8f*)(g_sh + (3 * 256 + j) * 16 + rb);
#pragma unroll
        for (int rr = 0; rr < 8; ++rr) {
            const int r = rb + rr;
            float i_ = fast_sigmoid(gi[rr] + b0s);
            float f_ = fast_sigmoid(gf[rr] + b1s);
            float g_ = fast_tanh   (gg[rr] + b2s);
            float o_ = fast_sigmoid(go[rr] + b3s);
            float c  = f_ * c_reg[rr] + i_ * g_;
            c_reg[rr] = c;
            float h  = o_ * fast_tanh(c);
            h_reg[rr] = h;
            // write h (bf16) into A's h-half for next step's GEMM
            a_sh[(kt_h * 32 + (r + lane_hi)) * 16 + slot_c] = f2bf(h);
        }
        __syncthreads();
    }

    // ---- emit (h_T, c_T), f32, concatenated
#pragma unroll
    for (int rr = 0; rr < 8; ++rr) {
        const int r = rb + rr;
        out[(size_t)(n0 + r) * H_DIM + j] = h_reg[rr];
        out[(size_t)N_BATCH * H_DIM + (size_t)(n0 + r) * H_DIM + j] = c_reg[rr];
    }
}

// ---------------------------------------------------------------------------
extern "C" void kernel_launch(void* const* d_in, const int* in_sizes, int n_in,
                              void* d_out, int out_size, void* d_ws, size_t ws_size,
                              hipStream_t stream)
{
    const float* x   = (const float*)d_in[0];
    const float* wii = (const float*)d_in[1];
    const float* whi = (const float*)d_in[2];
    const float* wif = (const float*)d_in[3];
    const float* whf = (const float*)d_in[4];
    const float* wig = (const float*)d_in[5];
    const float* whg = (const float*)d_in[6];
    const float* wio = (const float*)d_in[7];
    const float* who = (const float*)d_in[8];
    const float* bii = (const float*)d_in[9];
    const float* bhi = (const float*)d_in[10];
    const float* bif = (const float*)d_in[11];
    const float* bhf = (const float*)d_in[12];
    const float* big = (const float*)d_in[13];
    const float* bhg = (const float*)d_in[14];
    const float* bio = (const float*)d_in[15];
    const float* bho = (const float*)d_in[16];

    uint16_t* wpack = (uint16_t*)d_ws;                                   // 1 MB
    float*    bias  = (float*)((char*)d_ws + (size_t)K_TOT * G4H * 2);   // 4 KB

    pack_weights<<<(K_TOT * G4H + 255) / 256, 256, 0, stream>>>(
        wii, whi, wif, whf, wig, whg, wio, who, wpack);
    pack_bias<<<(G4H + 255) / 256, 256, 0, stream>>>(
        bii, bhi, bif, bhf, big, bhg, bio, bho, bias);

    const int smem_bytes = KT_N * 32 * 16 * 2 + 16 * G4H * 4;  // 16 KB + 64 KB
    hipFuncSetAttribute(reinterpret_cast<const void*>(lstm_persistent),
                        hipFuncAttributeMaxDynamicSharedMemorySize, smem_bytes);
    lstm_persistent<<<N_BATCH / 16, 512, smem_bytes, stream>>>(
        x, wpack, bias, (float*)d_out);
}
